// CoAttention_DNSImage_13391708029217
// MI455X (gfx1250) — compile-verified, matrix operand order checked by reading, not run on hardware
//
#include <hip/hip_runtime.h>
#include <math.h>

#define B 64
#define S 256
#define R 196
#define H 1024

typedef __bf16 bf16;
typedef __attribute__((ext_vector_type(16))) bf16  v16bf;
typedef __attribute__((ext_vector_type(8)))  bf16  v8bf;
typedef __attribute__((ext_vector_type(8)))  float v8f;

// Hardware tanh if the toolchain exposes it (gfx1250 v_tanh_f32), else libm.
__device__ __forceinline__ float fast_tanh(float x) {
#if __has_builtin(__builtin_amdgcn_tanhf)
  return __builtin_amdgcn_tanhf(x);
#elif __has_builtin(__builtin_amdgcn_tanh_f32)
  return __builtin_amdgcn_tanh_f32(x);
#else
  return tanhf(x);
#endif
}

// ---------------------------------------------------------------------------
// Kernel 1: convert fp32 weight matrices (H x H, row-major, out = x @ W^T) into
// bf16 WMMA B-operand fragments. Fragment layout for v_wmma_f32_16x16x32_bf16:
//   lane l holds column n = l&15 of the 16-wide N-chunk,
//   slot j -> k = (l>>4)*8 + (j&7) + (j>>3)*16 within the 32-wide K-chunk.
// Stored as [64 nchunk][32 kchunk][32 lane][16 bf16] -> contiguous 1KB per frag,
// so the GEMM kernel's per-lane 32B load is perfectly coalesced.
// ---------------------------------------------------------------------------
__global__ void wconvert_kernel(const float* __restrict__ W1,
                                const float* __restrict__ W2,
                                bf16* __restrict__ F1,
                                bf16* __restrict__ F2) {
  unsigned gid = blockIdx.x * 256u + threadIdx.x;   // 2 * 2^20 total
  unsigned mat = gid >> 20;
  unsigned e   = gid & 0xFFFFFu;
  unsigned j   = e & 15u;
  unsigned l   = (e >> 4) & 31u;
  unsigned kc  = (e >> 9) & 31u;
  unsigned nc  = e >> 14;
  unsigned n = nc * 16u + (l & 15u);
  unsigned k = kc * 32u + (l >> 4) * 8u + (j & 7u) + ((j >> 3) * 16u);
  const float* W = mat ? W2 : W1;
  bf16*        F = mat ? F2 : F1;
  F[e] = (bf16)W[n * 1024u + k];
}

// ---------------------------------------------------------------------------
// Kernel 2: fused tanh-GEMM reduced to one scalar per row.
//   out[row] = sum_N tanh( X[row,:] . W[N,:] + bias[N] ) * wv[N]
// One workgroup = 64 rows (4 tiles of 16) staged in LDS pre-packed in the
// WMMA A-fragment layout [kc][lane][16 bf16] (32KB/tile, bank-conflict-free
// 32B-per-lane reads). 8 waves x 4 pair-iterations cover the 64 N-chunks,
// TWO N-chunks in flight per wave: each A-fragment LDS load feeds 2 WMMAs,
// each B-fragment global load feeds 4 WMMAs (1.0 ds + 0.5 vmem per WMMA).
// ---------------------------------------------------------------------------
__global__ void scores_kernel(const float* __restrict__ dns,
                              const float* __restrict__ img,
                              const bf16* __restrict__ F1,
                              const bf16* __restrict__ F2,
                              const float* __restrict__ b1,
                              const float* __restrict__ b2,
                              const float* __restrict__ wa1,
                              const float* __restrict__ wa2,
                              float* __restrict__ s_dns,
                              float* __restrict__ t_img) {
  // 4 tiles x [32 kc][32 lane][16 bf16] = 128 KB (CDNA5: 320 KB LDS/WG).
  __shared__ __align__(16) bf16 PA[4][32 * 32 * 16];
  __shared__ float slds[64];

  const int t   = threadIdx.x;
  const int blk = blockIdx.x;

  const float* X; const float* bias; const float* wv;
  const bf16* F; float* out;
  if (blk < (B * S) / 64) {                 // 256 dns blocks
    X = dns + (size_t)blk * 64 * H;
    bias = b1; wv = wa1 + H; F = F1; out = s_dns + blk * 64;
  } else {                                  // 196 img blocks
    const int bi = blk - (B * S) / 64;
    X = img + (size_t)bi * 64 * H;
    bias = b2; wv = wa2 + H; F = F2; out = t_img + bi * 64;
  }

  if (t < 64) slds[t] = 0.f;

  // Stage 64 rows (fp32) into LDS as bf16, pre-swizzled into A-fragment order.
  for (int tt = 0; tt < 4; ++tt) {
    const float* Xt = X + (size_t)tt * 16 * H;
    for (int i = 0; i < 64; ++i) {
      const int e   = i * 256 + t;          // 0 .. 16383
      const int kc  = e >> 9;
      const int l   = (e >> 4) & 31;
      const int j   = e & 15;
      const int row = l & 15;
      const int k   = kc * 32 + ((l >> 4) << 3) + (j & 7) + ((j >> 3) << 4);
      PA[tt][e] = (bf16)Xt[row * H + k];
    }
  }
  __syncthreads();

  const int wave = t >> 5;
  const int lane = t & 31;
  const int half = lane >> 4;
  const int ln   = lane & 15;

  float racc[4][8];
#pragma unroll
  for (int tt = 0; tt < 4; ++tt)
#pragma unroll
    for (int j = 0; j < 8; ++j) racc[tt][j] = 0.f;

  // Wave owns N-chunks [wave*8, wave*8+8), processed as 4 pairs.
  for (int pi = 0; pi < 4; ++pi) {
    const int nc0 = wave * 8 + pi * 2;
    const int N0  = nc0 * 16 + ln;
    const int N1  = N0 + 16;
    const float bN0 = bias[N0], wN0 = wv[N0];
    const float bN1 = bias[N1], wN1 = wv[N1];

    const bf16* Fp = F + (((size_t)nc0 * 32) << 9);   // frag base for nc0

    v8f c00 = {0.f, 0.f, 0.f, 0.f, 0.f, 0.f, 0.f, 0.f};
    v8f c01 = c00, c02 = c00, c03 = c00;
    v8f c10 = c00, c11 = c00, c12 = c00, c13 = c00;

#pragma unroll 2
    for (int kc = 0; kc < 32; ++kc) {
      // Two B fragments (nc0, nc0+1): coalesced 32B/lane each.
      const v16bf bf0 = *(const v16bf*)(Fp + ((kc * 32 + lane) << 4));
      const v16bf bf1 = *(const v16bf*)(Fp + ((kc * 32 + lane) << 4) + (32 * 32 * 16));
      const int ao = ((kc * 32) + lane) << 4;         // 32B/lane, conflict-free
      const v16bf a0 = *(const v16bf*)&PA[0][ao];
      const v16bf a1 = *(const v16bf*)&PA[1][ao];
      const v16bf a2 = *(const v16bf*)&PA[2][ao];
      const v16bf a3 = *(const v16bf*)&PA[3][ao];
      c00 = __builtin_amdgcn_wmma_f32_16x16x32_bf16(false, a0, false, bf0, (short)0, c00, false, false);
      c10 = __builtin_amdgcn_wmma_f32_16x16x32_bf16(false, a0, false, bf1, (short)0, c10, false, false);
      c01 = __builtin_amdgcn_wmma_f32_16x16x32_bf16(false, a1, false, bf0, (short)0, c01, false, false);
      c11 = __builtin_amdgcn_wmma_f32_16x16x32_bf16(false, a1, false, bf1, (short)0, c11, false, false);
      c02 = __builtin_amdgcn_wmma_f32_16x16x32_bf16(false, a2, false, bf0, (short)0, c02, false, false);
      c12 = __builtin_amdgcn_wmma_f32_16x16x32_bf16(false, a2, false, bf1, (short)0, c12, false, false);
      c03 = __builtin_amdgcn_wmma_f32_16x16x32_bf16(false, a3, false, bf0, (short)0, c03, false, false);
      c13 = __builtin_amdgcn_wmma_f32_16x16x32_bf16(false, a3, false, bf1, (short)0, c13, false, false);
    }

    // c[j] = row (j + 8*half) of the tile. Fold tanh + w-dot for both N cols.
#pragma unroll
    for (int j = 0; j < 8; ++j) {
      racc[0][j] += fast_tanh(c00[j] + bN0) * wN0 + fast_tanh(c10[j] + bN1) * wN1;
      racc[1][j] += fast_tanh(c01[j] + bN0) * wN0 + fast_tanh(c11[j] + bN1) * wN1;
      racc[2][j] += fast_tanh(c02[j] + bN0) * wN0 + fast_tanh(c12[j] + bN1) * wN1;
      racc[3][j] += fast_tanh(c03[j] + bN0) * wN0 + fast_tanh(c13[j] + bN1) * wN1;
    }
  }

  // Reduce over the 16 lanes holding the 16 N-columns (per half-wave).
#pragma unroll
  for (int m = 1; m <= 8; m <<= 1)
#pragma unroll
    for (int tt = 0; tt < 4; ++tt)
#pragma unroll
      for (int j = 0; j < 8; ++j)
        racc[tt][j] += __shfl_xor(racc[tt][j], m, 16);

  if (ln == 0) {
#pragma unroll
    for (int tt = 0; tt < 4; ++tt)
#pragma unroll
      for (int j = 0; j < 8; ++j)
        atomicAdd(&slds[tt * 16 + j + 8 * half], racc[tt][j]);
  }
  __syncthreads();
  if (t < 64) out[t] = slds[t];
}

// ---------------------------------------------------------------------------
// Kernel 3: per (batch, stage): softmax over the score row (shift-invariance
// already removed the r-dependent terms), weighted pooling over the source
// features, then broadcast-store the pooled H-vector to all R output rows.
//   stage 0 -> att_dns  (scores s_dns over S, pool dns)
//   stage 1 -> att_img  (scores t_img over R, pool img)
// ---------------------------------------------------------------------------
__global__ void attend_kernel(const float* __restrict__ dns,
                              const float* __restrict__ img,
                              const float* __restrict__ s_dns,
                              const float* __restrict__ t_img,
                              float* __restrict__ outp) {
  __shared__ float p[256];
  __shared__ float red[256];

  const int t     = threadIdx.x;
  const int b     = blockIdx.x & 63;
  const int stage = blockIdx.x >> 6;

  int L; const float* sc; const float* X; float* out;
  if (stage == 0) {
    L = S; sc = s_dns + b * S; X = dns + (size_t)b * S * H;
    out = outp + (size_t)B * R * H + (size_t)b * R * H;   // att_dns is 2nd output
  } else {
    L = R; sc = t_img + b * R; X = img + (size_t)b * R * H;
    out = outp + (size_t)b * R * H;                       // att_img is 1st output
  }

  const float v = (t < L) ? sc[t] : -1e30f;
  red[t] = v;
  __syncthreads();
  for (int st = 128; st > 0; st >>= 1) {
    if (t < st) red[t] = fmaxf(red[t], red[t + st]);
    __syncthreads();
  }
  const float m = red[0];
  __syncthreads();
  const float e = (t < L) ? __expf(v - m) : 0.f;
  red[t] = e;
  __syncthreads();
  for (int st = 128; st > 0; st >>= 1) {
    if (t < st) red[t] += red[t + st];
    __syncthreads();
  }
  const float inv = 1.f / red[0];
  p[t] = e * inv;
  __syncthreads();

  // Pool: thread t owns h = 4t..4t+3 (float4, coalesced).
  float4 acc = {0.f, 0.f, 0.f, 0.f};
  const float4* Xv = (const float4*)X;
  for (int s = 0; s < L; ++s) {
    const float  ps = p[s];
    const float4 x  = Xv[s * (H / 4) + t];
    acc.x = fmaf(ps, x.x, acc.x);
    acc.y = fmaf(ps, x.y, acc.y);
    acc.z = fmaf(ps, x.z, acc.z);
    acc.w = fmaf(ps, x.w, acc.w);
  }

  // Broadcast to all R rows.
  float4* Ov = (float4*)out;
  for (int r = 0; r < R; ++r)
    Ov[r * (H / 4) + t] = acc;
}

// ---------------------------------------------------------------------------
extern "C" void kernel_launch(void* const* d_in, const int* in_sizes, int n_in,
                              void* d_out, int out_size, void* d_ws, size_t ws_size,
                              hipStream_t stream) {
  (void)in_sizes; (void)n_in; (void)out_size; (void)ws_size;

  const float* dns    = (const float*)d_in[0];
  const float* img    = (const float*)d_in[1];
  const float* W_dns1 = (const float*)d_in[2];
  const float* b_dns1 = (const float*)d_in[3];
  const float* w_att1 = (const float*)d_in[5];
  const float* W_img2 = (const float*)d_in[8];
  const float* b_img2 = (const float*)d_in[9];
  const float* w_att2 = (const float*)d_in[10];

  char* ws = (char*)d_ws;
  bf16*  F1 = (bf16*)ws;                     // 2 MB bf16 fragments
  bf16*  F2 = (bf16*)(ws + (1u << 21));      // 2 MB
  float* sd = (float*)(ws + (2u << 21));     // B*S scores (64 KB)
  float* ti = sd + B * S;                    // B*R scores (49 KB)

  wconvert_kernel<<<8192, 256, 0, stream>>>(W_dns1, W_img2, F1, F2);

  const int nblocks = (B * S) / 64 + (B * R) / 64;   // 256 + 196 = 452
  scores_kernel<<<nblocks, 256, 0, stream>>>(dns, img, F1, F2,
                                             b_dns1, b_img2, w_att1, w_att2,
                                             sd, ti);

  attend_kernel<<<2 * B, 256, 0, stream>>>(dns, img, sd, ti, (float*)d_out);
}
